// TemporalBlock_2499670966467
// MI455X (gfx1250) — compile-verified, matrix operand order checked by reading, not run on hardware
//
#include <hip/hip_runtime.h>
#include <math.h>

// ---------------------------------------------------------------------------
// TemporalBlock for MI455X (gfx1250), wave32.
//   - GEMMs: v_wmma_f32_16x16x32_f16 with LDS-staged tiles filled by
//     global_load_async_to_lds_b128 (ASYNCcnt) with double buffering.
//   - Problem is HBM-bound (~580 GFLOP vs ~0.5 GB traffic @ 23.3 TB/s), so
//     bias/GELU/residual/transpose are fused into GEMM epilogues.
// ---------------------------------------------------------------------------

typedef __attribute__((ext_vector_type(16))) _Float16 v16h;
typedef __attribute__((ext_vector_type(8)))  _Float16 v8h;
typedef __attribute__((ext_vector_type(8)))  float    v8f;

#define B_   2
#define T_   5
#define E_   4096
#define C_   768
#define H_   12
#define MT_  5
#define HID_ 3072
#define M_ROWS (B_*E_*T_)   // 40960 rows for every GEMM

#define SSTR 40             // LDS tile row stride in halves (80B: bank spread,
                            // still 16B-aligned for every b128 chunk)

union FragU { v16h v; v8h h[2]; };

// Async copy of 16 bytes global -> LDS (tracked by ASYNCcnt).
__device__ inline void async_cp_b128(unsigned lds_off, const void* gaddr) {
  asm volatile("global_load_async_to_lds_b128 %0, %1, off"
               :: "v"(lds_off), "v"(gaddr) : "memory");
}
__device__ inline unsigned lds_addr(const void* p) {
  return (unsigned)(unsigned long long)(uintptr_t)p;   // low 32b = LDS offset
}

// WMMA 16-bit A/B fragment from an LDS tile (row-major, SSTR-half stride).
// Lane l = (m%16) + 16*g holds K = g*8 + {0..7} in VGPR0-3 and
// K = 16 + g*8 + {0..7} in VGPR4-7  => two contiguous 16B DS loads per lane.
__device__ inline v16h lds_frag(const _Float16* tile, int row0, int lane) {
  const int r = lane & 15;
  const int g = lane >> 4;
  const _Float16* p = tile + (row0 + r) * SSTR + g * 8;
  FragU u;
  u.h[0] = *reinterpret_cast<const v8h*>(p);
  u.h[1] = *reinterpret_cast<const v8h*>(p + 16);
  return u.v;
}

// ---------------------------------------------------------------------------
// Tiled WMMA GEMM:  out[M,N] = A[M,K] * W[N,K]^T  (+ fused epilogue)
// Block = 256 threads = 8 waves (2 M x 4 N); wave tile 64x32, block 128x128.
// K-step 32, LDS double-buffered via async global->LDS copies.
// MODE 0: fp16 out = acc + bias                          (QKV)
// MODE 1: fp32 out = acc + bias + residual               (proj + skip)
// MODE 2: fp16 out = gelu(acc + bias)                    (FC1)
// MODE 3: fp32 out = acc + bias + residual, transposed   (FC2 -> d_out)
// ---------------------------------------------------------------------------
template<int MODE>
__global__ __launch_bounds__(256) void gemm_wmma(
    const _Float16* __restrict__ A, const _Float16* __restrict__ W,
    const float* __restrict__ bias, const float* __restrict__ res,
    _Float16* __restrict__ outH, float* __restrict__ outF,
    int M, int N, int K)
{
  __shared__ _Float16 sA[2][128 * SSTR];
  __shared__ _Float16 sB[2][128 * SSTR];

  const int tid  = threadIdx.x;
  const int lane = tid & 31;
  const int wave = tid >> 5;
  const int wm   = wave >> 2;          // 0..1
  const int wn   = wave & 3;           // 0..3
  const int mblk = blockIdx.y * 128;
  const int nblk = blockIdx.x * 128;

  // Staging: 128x32 halves = 8KB per tile = 512 x 16B chunks; each thread
  // copies chunk tid (rows 0..63) and tid+256 (rows 64..127) of A and of B.
  const int r0 = tid >> 2;             // 0..63
  const int r1 = r0 + 64;              // 64..127
  const int kc = (tid & 3) * 8;        // half offset within k-slab

  const _Float16* gA0 = A + (size_t)(mblk + r0) * K + kc;
  const _Float16* gA1 = A + (size_t)(mblk + r1) * K + kc;
  const _Float16* gB0 = W + (size_t)(nblk + r0) * K + kc;
  const _Float16* gB1 = W + (size_t)(nblk + r1) * K + kc;

  auto issue_stage = [&](int buf, int kk) {
    async_cp_b128(lds_addr(&sA[buf][r0 * SSTR + kc]), gA0 + kk);
    async_cp_b128(lds_addr(&sA[buf][r1 * SSTR + kc]), gA1 + kk);
    async_cp_b128(lds_addr(&sB[buf][r0 * SSTR + kc]), gB0 + kk);
    async_cp_b128(lds_addr(&sB[buf][r1 * SSTR + kc]), gB1 + kk);
  };

  v8f acc[4][2] = {};

  issue_stage(0, 0);                   // prologue: stage k=0 into buffer 0
  int buf = 0;
  for (int k0 = 0; k0 < K; k0 += 32) {
    const bool more = (k0 + 32) < K;
    if (more) {
      issue_stage(buf ^ 1, k0 + 32);   // prefetch next slab
      asm volatile("s_wait_asynccnt 0x4" ::: "memory");  // current slab done
    } else {
      asm volatile("s_wait_asynccnt 0x0" ::: "memory");
    }
    __syncthreads();                   // all waves' copies visible

    v16h a[4], b[2];
#pragma unroll
    for (int i = 0; i < 4; ++i) a[i] = lds_frag(sA[buf], wm * 64 + 16 * i, lane);
#pragma unroll
    for (int j = 0; j < 2; ++j) b[j] = lds_frag(sB[buf], wn * 32 + 16 * j, lane);
#pragma unroll
    for (int i = 0; i < 4; ++i)
#pragma unroll
      for (int j = 0; j < 2; ++j)
        acc[i][j] = __builtin_amdgcn_wmma_f32_16x16x32_f16(
            false, a[i], false, b[j], (short)0, acc[i][j], false, false);

    __syncthreads();                   // safe to overwrite buf next iteration
    buf ^= 1;
  }

  const int m0 = mblk + wm * 64;
  const int n0 = nblk + wn * 32;
  // C/D layout: VGPR r, lane l -> m_local = r + 8*(l>>4), n_local = l&15
#pragma unroll
  for (int i = 0; i < 4; ++i) {
#pragma unroll
    for (int j = 0; j < 2; ++j) {
#pragma unroll
      for (int r = 0; r < 8; ++r) {
        const int m = m0 + 16 * i + r + 8 * (lane >> 4);
        const int n = n0 + 16 * j + (lane & 15);
        float v = acc[i][j][r] + bias[n];
        if (MODE == 0) {
          outH[(size_t)m * N + n] = (_Float16)v;
        } else if (MODE == 1) {
          outF[(size_t)m * N + n] = v + res[(size_t)m * N + n];
        } else if (MODE == 2) {
          float g = 0.5f * v * (1.0f + erff(v * 0.70710678118654752f));
          outH[(size_t)m * N + n] = (_Float16)g;
        } else {
          float y = v + res[(size_t)m * N + n];
          // row m = (b*E+e)*T + t  ->  out[(b*T+t)*E + e][n]
          const int nseq = m / T_;
          const int t    = m % T_;
          const int bb   = nseq / E_;
          const int e    = nseq % E_;
          const size_t oidx = ((size_t)((bb * T_ + t) * E_) + e) * C_ + n;
          outF[oidx] = y;
        }
      }
    }
  }
}

// ---------------------------------------------------------------------------
// Row LayerNorm over C=768. One 256-thread block per row.
// FIRST=1: gather-transpose from x(B,T,E,C), save fp32 residual copy too.
// ---------------------------------------------------------------------------
template<int FIRST>
__global__ __launch_bounds__(256) void ln_kernel(
    const float* __restrict__ src, const float* __restrict__ w,
    const float* __restrict__ b,
    float* __restrict__ saveF, _Float16* __restrict__ outH)
{
  __shared__ float s1[256], s2[256];
  const int m   = blockIdx.x;
  const int tid = threadIdx.x;

  const float* row;
  if (FIRST) {
    const int nseq = m / T_;
    const int t    = m % T_;
    const int bb   = nseq / E_;
    const int e    = nseq % E_;
    row = src + ((size_t)((bb * T_ + t) * E_) + e) * C_;
  } else {
    row = src + (size_t)m * C_;
  }

  float a0 = 0.f, a1 = 0.f;
  for (int c = tid; c < C_; c += 256) { float x = row[c]; a0 += x; a1 += x * x; }
  s1[tid] = a0; s2[tid] = a1;
  __syncthreads();
  for (int off = 128; off > 0; off >>= 1) {
    if (tid < off) { s1[tid] += s1[tid + off]; s2[tid] += s2[tid + off]; }
    __syncthreads();
  }
  const float mean = s1[0] * (1.0f / C_);
  const float var  = s2[0] * (1.0f / C_) - mean * mean;
  const float rstd = rsqrtf(var + 1e-5f);

  for (int c = tid; c < C_; c += 256) {
    const float x = row[c];
    if (FIRST) saveF[(size_t)m * C_ + c] = x;
    outH[(size_t)m * C_ + c] = (_Float16)((x - mean) * rstd * w[c] + b[c]);
  }
}

// ---------------------------------------------------------------------------
// Attention: one 64-thread block (2 waves) per (sequence, head).
// T=5, dh=64: logits via wave32 butterfly reductions, bias from table,
// per-head scale exp(min(flex, log 100)), softmax, weighted V.
// ---------------------------------------------------------------------------
__global__ __launch_bounds__(64) void attn_kernel(
    const _Float16* __restrict__ qkv, const float* __restrict__ flex,
    const float* __restrict__ bias_table, const int* __restrict__ rel_index,
    _Float16* __restrict__ attn_out)
{
  const int nseq = blockIdx.x;
  const int h    = blockIdx.y;
  const int d    = threadIdx.x;          // 0..63 = head dim
  const int lane = d & 31;
  const int wv   = d >> 5;

  __shared__ float wred[2][25];
  __shared__ float logits[25];

  float q[T_], k[T_], vv[T_];
#pragma unroll
  for (int t = 0; t < T_; ++t) {
    const size_t base = ((size_t)(nseq * T_ + t)) * (3 * C_) + h * 64 + d;
    q[t]  = (float)qkv[base];
    k[t]  = (float)qkv[base + C_];
    vv[t] = (float)qkv[base + 2 * C_];
  }

#pragma unroll
  for (int t = 0; t < T_; ++t) {
#pragma unroll
    for (int s = 0; s < T_; ++s) {
      float p = q[t] * k[s];
#pragma unroll
      for (int off = 16; off > 0; off >>= 1) p += __shfl_xor(p, off, 32);
      if (lane == 0) wred[wv][t * T_ + s] = p;
    }
  }
  __syncthreads();

  const float scale = expf(fminf(flex[h], 4.6051701859880914f)); // log(100)
  if (d < 25) {
    const int t = d / T_, s = d % T_;
    logits[d] = (wred[0][d] + wred[1][d]) * 0.125f * scale   // 1/sqrt(64)
              + bias_table[rel_index[t * MT_ + s] * H_ + h];
  }
  __syncthreads();

#pragma unroll
  for (int t = 0; t < T_; ++t) {
    float l[T_], mx = -1e30f;
#pragma unroll
    for (int s = 0; s < T_; ++s) { l[s] = logits[t * T_ + s]; mx = fmaxf(mx, l[s]); }
    float sum = 0.f, o = 0.f;
#pragma unroll
    for (int s = 0; s < T_; ++s) { float e = expf(l[s] - mx); sum += e; o += e * vv[s]; }
    attn_out[((size_t)(nseq * T_ + t)) * C_ + h * 64 + d] = (_Float16)(o / sum);
  }
}

__global__ __launch_bounds__(256) void cvt_f2h(const float* __restrict__ s,
                                               _Float16* __restrict__ dst, int n)
{
  const int i = blockIdx.x * 256 + threadIdx.x;
  if (i < n) dst[i] = (_Float16)s[i];
}

// ---------------------------------------------------------------------------
extern "C" void kernel_launch(void* const* d_in, const int* in_sizes, int n_in,
                              void* d_out, int out_size, void* d_ws, size_t ws_size,
                              hipStream_t stream)
{
  (void)in_sizes; (void)n_in; (void)out_size; (void)ws_size;

  const float* x        = (const float*)d_in[0];
  const float* flex     = (const float*)d_in[1];
  const float* bias_tab = (const float*)d_in[2];
  const int*   rel_idx  = (const int*)d_in[3];
  const float* n1w      = (const float*)d_in[4];
  const float* n1b      = (const float*)d_in[5];
  const float* qkv_w    = (const float*)d_in[6];
  const float* qkv_b    = (const float*)d_in[7];
  const float* proj_w   = (const float*)d_in[8];
  const float* proj_b   = (const float*)d_in[9];
  const float* n2w      = (const float*)d_in[10];
  const float* n2b      = (const float*)d_in[11];
  const float* fc1_w    = (const float*)d_in[12];
  const float* fc1_b    = (const float*)d_in[13];
  const float* fc2_w    = (const float*)d_in[14];
  const float* fc2_b    = (const float*)d_in[15];
  float* out = (float*)d_out;

  // Workspace layout (deterministic, ~580 MB). The fp16 QKV + attn-out region
  // is reused for the FC1 hidden activations once attention/proj are done.
  char*  ws  = (char*)d_ws;
  size_t off = 0;
  auto alloc = [&](size_t bytes) -> void* {
    void* p = ws + off;
    off += (bytes + 255) & ~(size_t)255;
    return p;
  };

  float*    Xres = (float*)alloc((size_t)M_ROWS * C_ * sizeof(float));       // pre-norm residual
  float*    Y1   = (float*)alloc((size_t)M_ROWS * C_ * sizeof(float));       // x_ + proj out
  _Float16* XN   = (_Float16*)alloc((size_t)M_ROWS * C_ * sizeof(_Float16)); // ln1/ln2 rows
  _Float16* R    = (_Float16*)alloc((size_t)M_ROWS * HID_ * sizeof(_Float16));
  _Float16* QKV  = R;                                   // M x 2304 (fp16)
  _Float16* AOUT = R + (size_t)M_ROWS * (3 * C_);       // M x 768  (fp16)
  _Float16* HDN  = R;                                   // M x 3072 (reuse)
  _Float16* qkvWh  = (_Float16*)alloc((size_t)3 * C_ * C_ * sizeof(_Float16));
  _Float16* projWh = (_Float16*)alloc((size_t)C_ * C_ * sizeof(_Float16));
  _Float16* fc1Wh  = (_Float16*)alloc((size_t)HID_ * C_ * sizeof(_Float16));
  _Float16* fc2Wh  = (_Float16*)alloc((size_t)C_ * HID_ * sizeof(_Float16));

  int n;
  n = 3 * C_ * C_; cvt_f2h<<<(n + 255) / 256, 256, 0, stream>>>(qkv_w,  qkvWh,  n);
  n = C_ * C_;     cvt_f2h<<<(n + 255) / 256, 256, 0, stream>>>(proj_w, projWh, n);
  n = HID_ * C_;   cvt_f2h<<<(n + 255) / 256, 256, 0, stream>>>(fc1_w,  fc1Wh,  n);
  n = C_ * HID_;   cvt_f2h<<<(n + 255) / 256, 256, 0, stream>>>(fc2_w,  fc2Wh,  n);

  // 1) transpose + LN1
  ln_kernel<1><<<M_ROWS, 256, 0, stream>>>(x, n1w, n1b, Xres, XN);

  // 2) QKV = XN * qkv_w^T + b       (40960 x 2304 x 768)
  gemm_wmma<0><<<dim3((3 * C_) / 128, M_ROWS / 128), 256, 0, stream>>>(
      XN, qkvWh, qkv_b, nullptr, QKV, nullptr, M_ROWS, 3 * C_, C_);

  // 3) attention per (seq, head)
  attn_kernel<<<dim3(B_ * E_, H_), 64, 0, stream>>>(QKV, flex, bias_tab, rel_idx, AOUT);

  // 4) Y1 = AOUT * proj_w^T + b + Xres
  gemm_wmma<1><<<dim3(C_ / 128, M_ROWS / 128), 256, 0, stream>>>(
      AOUT, projWh, proj_b, Xres, nullptr, Y1, M_ROWS, C_, C_);

  // 5) LN2
  ln_kernel<0><<<M_ROWS, 256, 0, stream>>>(Y1, n2w, n2b, nullptr, XN);

  // 6) HDN = gelu(XN * fc1_w^T + b)   (40960 x 3072 x 768)
  gemm_wmma<2><<<dim3(HID_ / 128, M_ROWS / 128), 256, 0, stream>>>(
      XN, fc1Wh, fc1_b, nullptr, HDN, nullptr, M_ROWS, HID_, C_);

  // 7) out = transpose(HDN * fc2_w^T + b + Y1)   (40960 x 768 x 3072)
  gemm_wmma<3><<<dim3(C_ / 128, M_ROWS / 128), 256, 0, stream>>>(
      HDN, fc2Wh, fc2_b, Y1, nullptr, out, M_ROWS, C_, HID_);
}
